// GAT_63324997812472
// MI455X (gfx1250) — matrix operand order repaired and checked by reading, not compiled, not run on hardware
//
#include <hip/hip_runtime.h>
#include <hip/hip_bf16.h>
#include <stddef.h>

#define NNODES     10000
#define IN_DIM     10000
#define HIDF       128
#define NCLS       3
#define NEDGE      320000
#define ETOT       (NEDGE + NNODES)   // with self loops
#define NEG_SLOPE  0.2f

typedef __attribute__((ext_vector_type(16))) __bf16 bfx16;
typedef __attribute__((ext_vector_type(8)))  __bf16 bfx8;
typedef __attribute__((ext_vector_type(4)))  __bf16 bfx4;
typedef __attribute__((ext_vector_type(8)))  float  fx8;
typedef __attribute__((ext_vector_type(4)))  float  fx4;

// ---------------------------------------------------------------------------
// GEMM1: h1[N,128] = x[N,10000] @ W1[10000,128], bf16 WMMA, fp32 accumulate.
// Block: 256 thr (8 waves). M-tile 64, N full 128, K-step 64.
// wave w: mstrip=w&3, nhalf=w>>2 -> 4 C-tiles of 16x16 per wave,
// 8 WMMAs per wave per K-step. Double-buffered LDS, register-staged
// global loads one K-tile ahead (software pipeline, 1 barrier/iter).
// ---------------------------------------------------------------------------
__global__ __launch_bounds__(256) void gat_gemm1(const float* __restrict__ x,
                                                 const float* __restrict__ W1,
                                                 float* __restrict__ h1) {
    constexpr int KSTEP = 64;
    constexpr int LDK   = 72;    // padded bf16 row stride (144 B, 16B aligned)
    __shared__ __align__(16) __bf16 lA[2][64  * LDK];  // [row][k]
    __shared__ __align__(16) __bf16 lB[2][128 * LDK];  // [n][k] (W1 transposed)

    const int tid    = threadIdx.x;
    const int wave   = tid >> 5;
    const int lane   = tid & 31;
    const int half   = lane >> 4;
    const int l15    = lane & 15;
    const int mstrip = wave & 3;
    const int nhalf  = wave >> 2;
    const int m0     = blockIdx.x * 64;

    fx8 acc[4] = {};
    fx4 sA[4];            // staged A: 4 tasks  (64 rows x 16 chunks / 256 thr)
    fx4 sB[8];            // staged B: 8 tasks  (128 n  x 16 chunks / 256 thr)

    // ---- stage loaders (global -> regs) ----
    auto load_stage = [&](int k0) {
        #pragma unroll
        for (int i = 0; i < 4; ++i) {
            int tk = tid + i * 256;
            int r = tk >> 4, c = tk & 15;
            int gr = m0 + r, gk = k0 + c * 4;
            fx4 v = {};
            if (gr < NNODES && gk + 4 <= IN_DIM)
                v = *(const fx4*)(x + (size_t)gr * IN_DIM + gk);
            sA[i] = v;
        }
        #pragma unroll
        for (int i = 0; i < 8; ++i) {
            int tk = tid + i * 256;
            int n = tk & 127, c = tk >> 7;
            fx4 v = {};
            #pragma unroll
            for (int j = 0; j < 4; ++j) {
                int gk = k0 + c * 4 + j;
                if (gk < IN_DIM) v[j] = W1[(size_t)gk * HIDF + n];
            }
            sB[i] = v;
        }
    };
    // ---- stage store (regs -> LDS, packed bf16, b64 stores) ----
    auto store_stage = [&](int buf) {
        #pragma unroll
        for (int i = 0; i < 4; ++i) {
            int tk = tid + i * 256;
            int r = tk >> 4, c = tk & 15;
            *(bfx4*)(&lA[buf][r * LDK + c * 4]) = __builtin_convertvector(sA[i], bfx4);
        }
        #pragma unroll
        for (int i = 0; i < 8; ++i) {
            int tk = tid + i * 256;
            int n = tk & 127, c = tk >> 7;
            *(bfx4*)(&lB[buf][n * LDK + c * 4]) = __builtin_convertvector(sB[i], bfx4);
        }
    };
    // ---- compute 8 WMMAs from LDS buffer ----
    auto compute = [&](int buf) {
        #pragma unroll
        for (int kb = 0; kb < KSTEP; kb += 32) {
            bfx16 a;
            const __bf16* ap =
                &lA[buf][(mstrip * 16 + l15) * LDK + kb + (half ? 8 : 0)];
            *((bfx8*)&a)     = *(const bfx8*)ap;
            *((bfx8*)&a + 1) = *(const bfx8*)(ap + 16);
            #pragma unroll
            for (int t = 0; t < 4; ++t) {
                int ncol = nhalf * 64 + t * 16 + l15;
                bfx16 b;
                const __bf16* bp = &lB[buf][ncol * LDK + kb + (half ? 8 : 0)];
                *((bfx8*)&b)     = *(const bfx8*)bp;
                *((bfx8*)&b + 1) = *(const bfx8*)(bp + 16);
                acc[t] = __builtin_amdgcn_wmma_f32_16x16x32_bf16(
                             false, a, false, b, (short)0, acc[t], false, false);
            }
        }
    };

    // ---- software-pipelined main loop ----
    load_stage(0);
    store_stage(0);
    __syncthreads();
    int cur = 0;
    const int nk = (IN_DIM + KSTEP - 1) / KSTEP;   // 157
    for (int ks = 0; ks < nk; ++ks) {
        int k0n = (ks + 1) * KSTEP;
        bool has_next = k0n < IN_DIM;
        if (has_next) load_stage(k0n);     // global loads in flight ...
        compute(cur);                      // ... while WMMAs run from LDS
        if (has_next) store_stage(cur ^ 1);
        __syncthreads();
        cur ^= 1;
    }

    // ---- store C: vgpr r, lanes0-15 -> M=r, lanes16-31 -> M=8+r ----
    #pragma unroll
    for (int t = 0; t < 4; ++t) {
        int col = nhalf * 64 + t * 16 + l15;
        #pragma unroll
        for (int r = 0; r < 8; ++r) {
            int row = m0 + mstrip * 16 + half * 8 + r;
            if (row < NNODES) h1[(size_t)row * HIDF + col] = acc[t][r];
        }
    }
}

// ---------------------------------------------------------------------------
// Per-node attention scores for layer 1 (wave per node, 128 features)
// ---------------------------------------------------------------------------
__global__ __launch_bounds__(256) void gat_alpha1(const float* __restrict__ h1,
                                                  const float* __restrict__ asrc,
                                                  const float* __restrict__ adst,
                                                  float* __restrict__ als,
                                                  float* __restrict__ ald) {
    int node = blockIdx.x * 8 + (threadIdx.x >> 5);
    int lane = threadIdx.x & 31;
    if (node >= NNODES) return;
    const float* hp = h1 + (size_t)node * HIDF;
    float s = 0.f, d = 0.f;
    for (int i = lane; i < HIDF; i += 32) {
        float v = hp[i];
        s += v * asrc[i];
        d += v * adst[i];
    }
    #pragma unroll
    for (int off = 16; off; off >>= 1) {
        s += __shfl_xor(s, off, 32);
        d += __shfl_xor(d, off, 32);
    }
    if (lane == 0) { als[node] = s; ald[node] = d; }
}

__device__ __forceinline__ void edge_sd(const int* ei, int idx, int& s, int& d) {
    if (idx < NEDGE) { s = ei[idx]; d = ei[NEDGE + idx]; }
    else             { s = d = idx - NEDGE; }          // self loop
}

// monotone uint encoding of float for atomicMax
__device__ __forceinline__ unsigned fkey(float f) {
    unsigned u = __float_as_uint(f);
    return (u & 0x80000000u) ? ~u : (u | 0x80000000u);
}
__device__ __forceinline__ float funkey(unsigned k) {
    return __uint_as_float((k & 0x80000000u) ? (k ^ 0x80000000u) : ~k);
}

// e = leaky_relu(als[src]+ald[dst]); segment max over dst via atomicMax(key)
__global__ __launch_bounds__(256) void gat_edge_max(const int* __restrict__ ei,
                                                    const float* __restrict__ als,
                                                    const float* __restrict__ ald,
                                                    float* __restrict__ e,
                                                    unsigned* __restrict__ maxk) {
    int idx = blockIdx.x * 256 + threadIdx.x;
    if (idx >= ETOT) return;
    int s, d; edge_sd(ei, idx, s, d);
    float v = als[s] + ald[d];
    v = (v > 0.f) ? v : NEG_SLOPE * v;
    e[idx] = v;
    atomicMax(maxk + d, fkey(v));
}

// e <- exp(e - max[dst]);  denom[dst] += e
__global__ __launch_bounds__(256) void gat_edge_exp(const int* __restrict__ ei,
                                                    float* __restrict__ e,
                                                    const unsigned* __restrict__ maxk,
                                                    float* __restrict__ denom) {
    int idx = blockIdx.x * 256 + threadIdx.x;
    if (idx >= ETOT) return;
    int s, d; edge_sd(ei, idx, s, d);
    float p = __expf(e[idx] - funkey(maxk[d]));
    e[idx] = p;
    atomicAdd(denom + d, p);
}

// layer-1 scatter: agg[dst] += h1[src] * attn  (32 threads/edge, 4 feats each)
__global__ __launch_bounds__(256) void gat_scatter1(const int* __restrict__ ei,
                                                    const float* __restrict__ h1,
                                                    const float* __restrict__ p,
                                                    const float* __restrict__ denom,
                                                    float* __restrict__ agg) {
    long t = (long)blockIdx.x * 256 + threadIdx.x;
    int edge = (int)(t >> 5);
    if (edge >= ETOT) return;
    int fq = ((int)t & 31) * 4;
    int s, d; edge_sd(ei, edge, s, d);
    float attn = p[edge] / denom[d];
    const float4 hv = *(const float4*)(h1 + (size_t)s * HIDF + fq);
    float* ap = agg + (size_t)d * HIDF + fq;
    atomicAdd(ap + 0, hv.x * attn);
    atomicAdd(ap + 1, hv.y * attn);
    atomicAdd(ap + 2, hv.z * attn);
    atomicAdd(ap + 3, hv.w * attn);
}

// finish layer1 (bias+relu), project 128->3, layer-2 alpha scores (wave/node)
__global__ __launch_bounds__(256) void gat_node_l2(const float* __restrict__ agg,
                                                   const float* __restrict__ b1,
                                                   const float* __restrict__ W2,
                                                   const float* __restrict__ as2,
                                                   const float* __restrict__ ad2,
                                                   float* __restrict__ h2,
                                                   float* __restrict__ als2,
                                                   float* __restrict__ ald2) {
    int node = blockIdx.x * 8 + (threadIdx.x >> 5);
    int lane = threadIdx.x & 31;
    if (node >= NNODES) return;
    float c0 = 0.f, c1 = 0.f, c2 = 0.f;
    const float* ap = agg + (size_t)node * HIDF;
    for (int i = lane; i < HIDF; i += 32) {
        float v = ap[i] + b1[i];
        v = (v > 0.f) ? v : 0.f;                  // relu after layer 1
        c0 += v * W2[i * 3 + 0];
        c1 += v * W2[i * 3 + 1];
        c2 += v * W2[i * 3 + 2];
    }
    #pragma unroll
    for (int off = 16; off; off >>= 1) {
        c0 += __shfl_xor(c0, off, 32);
        c1 += __shfl_xor(c1, off, 32);
        c2 += __shfl_xor(c2, off, 32);
    }
    if (lane == 0) {
        h2[node * 3 + 0] = c0;
        h2[node * 3 + 1] = c1;
        h2[node * 3 + 2] = c2;
        als2[node] = c0 * as2[0] + c1 * as2[1] + c2 * as2[2];
        ald2[node] = c0 * ad2[0] + c1 * ad2[1] + c2 * ad2[2];
    }
}

// layer-2 scatter: acc2[dst][0..2] += h2[src][0..2] * attn  (thread per edge)
__global__ __launch_bounds__(256) void gat_scatter2(const int* __restrict__ ei,
                                                    const float* __restrict__ h2,
                                                    const float* __restrict__ p,
                                                    const float* __restrict__ denom,
                                                    float* __restrict__ acc2) {
    int idx = blockIdx.x * 256 + threadIdx.x;
    if (idx >= ETOT) return;
    int s, d; edge_sd(ei, idx, s, d);
    float attn = p[idx] / denom[d];
    atomicAdd(acc2 + (size_t)d * 3 + 0, h2[(size_t)s * 3 + 0] * attn);
    atomicAdd(acc2 + (size_t)d * 3 + 1, h2[(size_t)s * 3 + 1] * attn);
    atomicAdd(acc2 + (size_t)d * 3 + 2, h2[(size_t)s * 3 + 2] * attn);
}

__global__ __launch_bounds__(256) void gat_final(const float* __restrict__ acc2,
                                                 const float* __restrict__ b2,
                                                 float* __restrict__ out) {
    int idx = blockIdx.x * 256 + threadIdx.x;
    if (idx >= NNODES * NCLS) return;
    out[idx] = acc2[idx] + b2[idx % 3];
}

// ---------------------------------------------------------------------------
extern "C" void kernel_launch(void* const* d_in, const int* in_sizes, int n_in,
                              void* d_out, int out_size, void* d_ws, size_t ws_size,
                              hipStream_t stream) {
    const float* x    = (const float*)d_in[0];
    const int*   ei   = (const int*)  d_in[1];
    const float* W1   = (const float*)d_in[2];
    const float* as1  = (const float*)d_in[3];
    const float* ad1  = (const float*)d_in[4];
    const float* b1   = (const float*)d_in[5];
    const float* W2   = (const float*)d_in[6];
    const float* as2  = (const float*)d_in[7];
    const float* ad2  = (const float*)d_in[8];
    const float* b2   = (const float*)d_in[9];
    float* out = (float*)d_out;
    float* ws  = (float*)d_ws;

    // ---- workspace layout (floats). zero-init block first. ----
    size_t o = 0;
    float*    agg1   = ws + o;              o += (size_t)NNODES * HIDF;
    unsigned* maxk1  = (unsigned*)(ws + o); o += NNODES;
    float*    denom1 = ws + o;              o += NNODES;
    unsigned* maxk2  = (unsigned*)(ws + o); o += NNODES;
    float*    denom2 = ws + o;              o += NNODES;
    float*    acc2   = ws + o;              o += (size_t)NNODES * NCLS;
    size_t zero_bytes = o * sizeof(float);
    float*    h1   = ws + o;  o += (size_t)NNODES * HIDF;
    float*    als1 = ws + o;  o += NNODES;
    float*    ald1 = ws + o;  o += NNODES;
    float*    e1   = ws + o;  o += ETOT;
    float*    h2   = ws + o;  o += (size_t)NNODES * NCLS;
    float*    als2 = ws + o;  o += NNODES;
    float*    ald2 = ws + o;  o += NNODES;
    float*    e2   = ws + o;  o += ETOT;

    hipMemsetAsync(d_ws, 0, zero_bytes, stream);

    const int eb = (ETOT + 255) / 256;

    // layer 1
    gat_gemm1<<<(NNODES + 63) / 64, 256, 0, stream>>>(x, W1, h1);
    gat_alpha1<<<(NNODES + 7) / 8, 256, 0, stream>>>(h1, as1, ad1, als1, ald1);
    gat_edge_max<<<eb, 256, 0, stream>>>(ei, als1, ald1, e1, maxk1);
    gat_edge_exp<<<eb, 256, 0, stream>>>(ei, e1, maxk1, denom1);
    {
        long threads = (long)ETOT * 32;
        int blocks = (int)((threads + 255) / 256);
        gat_scatter1<<<blocks, 256, 0, stream>>>(ei, h1, e1, denom1, agg1);
    }
    // layer 2
    gat_node_l2<<<(NNODES + 7) / 8, 256, 0, stream>>>(agg1, b1, W2, as2, ad2,
                                                      h2, als2, ald2);
    gat_edge_max<<<eb, 256, 0, stream>>>(ei, als2, ald2, e2, maxk2);
    gat_edge_exp<<<eb, 256, 0, stream>>>(ei, e2, maxk2, denom2);
    gat_scatter2<<<eb, 256, 0, stream>>>(ei, h2, e2, denom2, acc2);
    gat_final<<<(NNODES * NCLS + 255) / 256, 256, 0, stream>>>(acc2, b2, out);
}
